// HEPOSEncoderLayer_75445395522269
// MI455X (gfx1250) — matrix operand order repaired and checked by reading, not compiled, not run on hardware
//
#include <hip/hip_runtime.h>
#include <hip/hip_bf16.h>
#include <math.h>
#include <stdint.h>

// ---------------------------------------------------------------------------
// HEPOS encoder layer for MI455X (gfx1250, wave32): WMMA bf16 GEMMs,
// TDM (tensor_load_to_lds) A-tile staging with LDS padding done by the DMA,
// async-to-LDS K staging in attention, flash-style online softmax.
// ---------------------------------------------------------------------------

typedef __bf16 bf16;
typedef __attribute__((ext_vector_type(16))) __bf16 v16bf;
typedef __attribute__((ext_vector_type(8)))  float  v8f;
typedef unsigned int u32x4 __attribute__((ext_vector_type(4)));
typedef int          i32x4 __attribute__((ext_vector_type(4)));
typedef int          i32x8 __attribute__((ext_vector_type(8)));

#define LN_EPSF 1e-5f

constexpr int Dm = 1024;   // model dim
constexpr int Bn = 2;      // batch
constexpr int Ln = 4096;   // seq len
constexpr int NH = 16;     // heads
constexpr int HD = 64;     // head dim

#if __has_builtin(__builtin_amdgcn_tensor_load_to_lds)
#define HAVE_TDM 1
#else
#define HAVE_TDM 0
#endif

__device__ __forceinline__ int head_stride(int h) {
  return (h < 2) ? 1 : (h < 6) ? 2 : (h < 10) ? 4 : (h < 14) ? 8 : 16;
}

// Async 16B global->LDS copy (CDNA5, ASYNCcnt-tracked). LDS byte address =
// low 32 bits of generic shared pointer.
__device__ __forceinline__ void async_copy_b128(const bf16* g, const bf16* lds) {
  unsigned l = (unsigned)(uintptr_t)lds;
  unsigned long long ga = (unsigned long long)(uintptr_t)g;
  asm volatile("global_load_async_to_lds_b128 %0, %1, off"
               :: "v"(l), "v"(ga) : "memory");
}
__device__ __forceinline__ void async_wait0() {
  asm volatile("s_wait_asynccnt 0x0" ::: "memory");
}

#if HAVE_TDM
// Issue one TDM 2D tile load: tile_h rows x tile_w elems (bf16) from a
// row-major tensor (row stride = row_stride elems) into LDS at lds_off,
// with hardware LDS padding: pad_amt_code+1 DWORDs every 2^(pad_int_code+1)
// DWORDs. D# layout per CDNA5 ISA 8.3/8.4. TDM ignores EXEC; TENSORcnt.
__device__ __forceinline__ void tdm_load_2d(unsigned lds_off, const void* gptr,
                                            unsigned tile_w, unsigned tile_h,
                                            unsigned tensor_w, unsigned tensor_h,
                                            unsigned row_stride,
                                            unsigned pad_int_code,
                                            unsigned pad_amt_code) {
  unsigned long long ga = (unsigned long long)(uintptr_t)gptr;
  u32x4 g0;
  g0.x = 1u;                                               // count=1, user D#
  g0.y = lds_off;                                          // lds_addr (bytes)
  g0.z = (unsigned)ga;                                     // global_addr lo
  g0.w = (unsigned)((ga >> 32) & 0x01FFFFFFull) | (2u << 30);  // hi | type=2
  i32x8 g1;
  g1[0] = (int)((1u << 16)                 // data_size = 2 bytes (bf16)
                | (1u << 20)               // pad_enable
                | (pad_int_code << 22)     // pad interval
                | (pad_amt_code << 25));   // pad amount
  g1[1] = (int)((tensor_w & 0xFFFFu) << 16);               // tensor_dim0 lo16
  g1[2] = (int)((tensor_w >> 16) | ((tensor_h & 0xFFFFu) << 16));
  g1[3] = (int)((tensor_h >> 16) | (tile_w << 16));        // tile_dim0
  g1[4] = (int)(tile_h & 0xFFFFu);                         // tile_dim1
  g1[5] = (int)row_stride;                                 // dim0_stride lo32
  g1[6] = 0;
  g1[7] = 0;
  i32x4 z4 = {0, 0, 0, 0};
#if defined(__clang_major__) && (__clang_major__ >= 23)
  i32x8 z8 = {0, 0, 0, 0, 0, 0, 0, 0};
  __builtin_amdgcn_tensor_load_to_lds(g0, g1, z4, z4, z8, 0);
#else
  __builtin_amdgcn_tensor_load_to_lds(g0, g1, z4, z4, 0);
#endif
}
#endif

// ---------------------------------------------------------------------------
// f32 -> bf16 conversion (grid-stride)
// ---------------------------------------------------------------------------
__global__ void f32_to_bf16_kernel(const float* __restrict__ in,
                                   bf16* __restrict__ out, int n) {
  int i = blockIdx.x * blockDim.x + threadIdx.x;
  int stride = gridDim.x * blockDim.x;
  for (; i < n; i += stride) out[i] = (bf16)in[i];
}

// ---------------------------------------------------------------------------
// Tiled GEMM: C[M,N] = act(A[M,K](bf16) @ B[K,N](bf16) + bias)
// BM=128, BN=128, BK=32. 8 waves (4x2); each wave = 32x64 tile = 8 WMMA/step.
// A tile staged by TDM into double-buffered LDS (DMA overlaps B-stage+WMMA);
// hardware pad gives 40-elem row stride. B staged transposed ([n][k]).
// ---------------------------------------------------------------------------
template <bool OUT_BF16, bool RELU>
__global__ __launch_bounds__(256) void gemm_bf16_kernel(
    const bf16* __restrict__ A, const bf16* __restrict__ Bw,
    const float* __restrict__ bias, void* __restrict__ Cout,
    int M, int N, int K) {
  constexpr int BM = 128, BN = 128, BK = 32;
  constexpr int SA = 40;          // As row stride: 16 DW row + 4 DW pad (TDM)
  constexpr int SB = 40;          // Bs row stride, Bs transposed [n][k]
  constexpr int ASZ = BM * SA;    // one A buffer (elems)
  __shared__ bf16 As[2 * ASZ];    // double-buffered A tile
  __shared__ bf16 Bs[BN * SB];

  const int tid  = threadIdx.x;
  const int lane = tid & 31;
  const int wave = tid >> 5;
  const int wm   = wave & 3;   // row wave group (0..3) -> 32 rows each
  const int wn   = wave >> 2;  // col wave group (0..1) -> 64 cols each
  const int l15  = lane & 15;
  const int hi   = lane >> 4;

  const int bm = blockIdx.y * BM;
  const int bn = blockIdx.x * BN;

  v8f acc[2][4] = {};

  union F { v16bf v; uint4 u[2]; };

  // ---- stage A tile k into buffer buf
  auto stage_A = [&](int k0, int buf) {
#if HAVE_TDM
    if (tid < 32)  // one DMA per block, issued from wave 0
      tdm_load_2d((unsigned)(uintptr_t)(As + buf * ASZ),
                  A + (size_t)bm * K + k0,
                  /*tile_w=*/BK, /*tile_h=*/BM,
                  /*tensor_w=*/(unsigned)K, /*tensor_h=*/(unsigned)M,
                  /*row_stride=*/(unsigned)K,
                  /*pad: every 16 DW*/ 3u, /*pad 4 DW*/ 3u);
#else
#pragma unroll
    for (int i = 0; i < 2; ++i) {
      int c   = tid + i * 256;
      int row = c >> 2;
      int kc  = (c & 3) * 8;
      async_copy_b128(A + (size_t)(bm + row) * K + k0 + kc,
                      As + buf * ASZ + row * SA + kc);
    }
#endif
  };
  auto wait_A = [&]() {
#if HAVE_TDM
    __builtin_amdgcn_s_wait_tensorcnt(0);
#else
    async_wait0();
#endif
  };

  stage_A(0, 0);

  int kb = 0;
  for (int k0 = 0; k0 < K; k0 += BK, kb ^= 1) {
    __syncthreads();  // all waves done reading previous tiles
    // ---- B tile [32][128] -> Bs[n][k] transposed scatter (2 chunks/thread)
#pragma unroll
    for (int i = 0; i < 2; ++i) {
      int c  = tid + i * 256;
      int kr = c >> 4;
      int nc = (c & 15) * 8;
      union { uint4 u; bf16 e[8]; } v;
      v.u = *(const uint4*)(Bw + (size_t)(k0 + kr) * N + bn + nc);
#pragma unroll
      for (int j = 0; j < 8; ++j) Bs[(nc + j) * SB + kr] = v.e[j];
    }
    wait_A();          // A(k0) DMA landed
    __syncthreads();   // publish As(k0) + Bs(k0)
    if (k0 + BK < K) stage_A(k0 + BK, kb ^ 1);  // overlap next A with compute

    const bf16* Ab = As + kb * ASZ;

    // ---- A fragments (2 row tiles)
    F a[2];
#pragma unroll
    for (int i = 0; i < 2; ++i) {
      int m  = wm * 32 + i * 16 + l15;
      int kk = hi ? 8 : 0;  // A 16-bit layout: lanes16-31 hold K=8..15,24..31
      a[i].u[0] = *(const uint4*)(Ab + m * SA + kk);
      a[i].u[1] = *(const uint4*)(Ab + m * SA + kk + 16);
    }
    // ---- 4 col tiles x 2 row tiles WMMA
#pragma unroll
    for (int j = 0; j < 4; ++j) {
      F b;
      int n  = wn * 64 + j * 16 + l15;
      int kk = hi ? 8 : 0;   // B 16-bit layout: lanes16-31 hold K=8..15,24..31
      b.u[0] = *(const uint4*)(Bs + n * SB + kk);
      b.u[1] = *(const uint4*)(Bs + n * SB + kk + 16);
#pragma unroll
      for (int i = 0; i < 2; ++i)
        acc[i][j] = __builtin_amdgcn_wmma_f32_16x16x32_bf16(
            false, a[i].v, false, b.v, (short)0, acc[i][j], false, false);
    }
  }

  // ---- epilogue: C layout => lane = column, VGPR = row
#pragma unroll
  for (int j = 0; j < 4; ++j) {
    int gc   = bn + wn * 64 + j * 16 + l15;
    float bv = bias ? bias[gc] : 0.f;
#pragma unroll
    for (int i = 0; i < 2; ++i) {
      int rbase = bm + wm * 32 + i * 16 + hi * 8;
#pragma unroll
      for (int v = 0; v < 8; ++v) {
        float x = acc[i][j][v] + bv;
        if (RELU) x = fmaxf(x, 0.f);
        size_t off = (size_t)(rbase + v) * N + gc;
        if (OUT_BF16) ((bf16*)Cout)[off] = (bf16)x;
        else          ((float*)Cout)[off] = x;
      }
    }
  }
}

// ---------------------------------------------------------------------------
// HEPOS attention: flash-style online softmax, strided K/V per head.
// Block = 128 query rows of one (b,h); 8 waves each own 16 rows.
// K tile staged via async-to-LDS; V staged transposed; P via LDS transpose.
// qkv layout: [B*L, 3072] bf16 (cols 0..1023=Q, 1024..2047=K, 2048..3071=V).
// ---------------------------------------------------------------------------
__global__ __launch_bounds__(256) void hepos_attn_kernel(
    const bf16* __restrict__ qkv, bf16* __restrict__ out) {
  constexpr int SK = 72;  // Ks row stride (144B, 16B multiple)
  constexpr int SV = 40;  // Vt row stride (80B)
  constexpr int SP = 40;  // P  row stride (80B)
  __shared__ bf16 Ks[32 * SK];      // [key][dim]
  __shared__ bf16 Vt[64 * SV];      // [dim][key] (transposed)
  __shared__ bf16 Pw[8 * 16 * SP];  // per-wave 16x32 probability tiles

  const int tid  = threadIdx.x;
  const int lane = tid & 31;
  const int wave = tid >> 5;
  const int l15  = lane & 15;
  const int hi   = lane >> 4;

  const int b  = blockIdx.z;
  const int h  = blockIdx.y;
  const int q0 = blockIdx.x * 128 + wave * 16;
  const int s  = head_stride(h);
  const int Lk = Ln / s;
  const float scale = 0.125f;  // 1/sqrt(64)

  const size_t rowstride = 3 * Dm;
  const bf16* qbase = qkv + (size_t)(b * Ln) * rowstride + h * HD;
  const bf16* kbase = qbase + Dm;
  const bf16* vbase = qbase + 2 * Dm;

  union F { v16bf v; uint4 u[2]; };

  // ---- resident Q fragments (head dim split into two K=32 steps)
  F aq[2];
  {
    const bf16* qrow = qbase + (size_t)(q0 + l15) * rowstride;
#pragma unroll
    for (int ks = 0; ks < 2; ++ks) {
      int d0 = ks * 32 + (hi ? 8 : 0);
      aq[ks].u[0] = *(const uint4*)(qrow + d0);
      aq[ks].u[1] = *(const uint4*)(qrow + d0 + 16);
    }
  }

  float mrow[8], lrow[8];
  v8f o[4] = {};
#pragma unroll
  for (int v = 0; v < 8; ++v) { mrow[v] = -3.0e38f; lrow[v] = 0.f; }

  bf16* Pme = Pw + wave * 16 * SP;

  for (int kt = 0; kt < Lk; kt += 32) {
    __syncthreads();
    // ---- stage K tile [32][64] (async) and V tile transposed [64][32]
    {
      int kk = tid >> 3;
      int dc = (tid & 7) * 8;
      size_t grow = (size_t)((kt + kk) * s) * rowstride;
      async_copy_b128(kbase + grow + dc, Ks + kk * SK + dc);
      union { uint4 u; bf16 e[8]; } vv;
      vv.u = *(const uint4*)(vbase + grow + dc);
#pragma unroll
      for (int j = 0; j < 8; ++j) Vt[(dc + j) * SV + kk] = vv.e[j];
    }
    async_wait0();
    __syncthreads();

    // ---- S = Q @ K^T : two 16-key subtiles, 2 WMMA each (K=64 head dim)
    v8f S[2];
#pragma unroll
    for (int ns = 0; ns < 2; ++ns) {
      v8f sacc = {};
#pragma unroll
      for (int ks = 0; ks < 2; ++ks) {
        F bk;
        int key = ns * 16 + l15;
        int d0  = ks * 32 + (hi ? 8 : 0);
        bk.u[0] = *(const uint4*)(Ks + key * SK + d0);
        bk.u[1] = *(const uint4*)(Ks + key * SK + d0 + 16);
        sacc = __builtin_amdgcn_wmma_f32_16x16x32_bf16(
            false, aq[ks].v, false, bk.v, (short)0, sacc, false, false);
      }
      S[ns] = sacc;
    }

    // ---- online softmax per query row (rows in VGPR idx, keys across lanes)
#pragma unroll
    for (int v = 0; v < 8; ++v) {
      float s0 = S[0][v] * scale, s1 = S[1][v] * scale;
      float mx = fmaxf(s0, s1);
#pragma unroll
      for (int d = 8; d >= 1; d >>= 1) mx = fmaxf(mx, __shfl_xor(mx, d, 16));
      float mnew  = fmaxf(mrow[v], mx);
      float alpha = __expf(mrow[v] - mnew);
      float p0 = __expf(s0 - mnew);
      float p1 = __expf(s1 - mnew);
      float rs = p0 + p1;
#pragma unroll
      for (int d = 8; d >= 1; d >>= 1) rs += __shfl_xor(rs, d, 16);
      lrow[v] = lrow[v] * alpha + rs;
      mrow[v] = mnew;
#pragma unroll
      for (int c = 0; c < 4; ++c) o[c][v] *= alpha;
      // C layout -> LDS P tile [row m][key]
      int m = v + hi * 8;
      Pme[m * SP + l15]      = (bf16)p0;  // keys 0..15
      Pme[m * SP + 16 + l15] = (bf16)p1;  // keys 16..31
    }

    // DS ops are in-order per wave; block compiler reordering + drain dscnt
    asm volatile("s_wait_dscnt 0x0" ::: "memory");

    // ---- O += P @ V  (A = 16x32 probs, B = V^T tiles from LDS)
    F ap;
    {
      int kk = hi ? 8 : 0;
      ap.u[0] = *(const uint4*)(Pme + l15 * SP + kk);
      ap.u[1] = *(const uint4*)(Pme + l15 * SP + kk + 16);
    }
#pragma unroll
    for (int c = 0; c < 4; ++c) {
      F bv;
      int d  = c * 16 + l15;
      int kk = hi ? 8 : 0;
      bv.u[0] = *(const uint4*)(Vt + d * SV + kk);
      bv.u[1] = *(const uint4*)(Vt + d * SV + kk + 16);
      o[c] = __builtin_amdgcn_wmma_f32_16x16x32_bf16(
          false, ap.v, false, bv.v, (short)0, o[c], false, false);
    }
  }

  // ---- normalize and write bf16 attention output [B*L, D] (head-concat)
  bf16* orow0 = out + (size_t)(b * Ln + q0) * Dm + h * HD;
#pragma unroll
  for (int c = 0; c < 4; ++c) {
#pragma unroll
    for (int v = 0; v < 8; ++v) {
      int m = v + hi * 8;
      float val = o[c][v] / lrow[v];
      orow0[(size_t)m * Dm + c * 16 + l15] = (bf16)val;
    }
  }
}

// ---------------------------------------------------------------------------
// out = LayerNorm(X + Y) * gamma + beta ; optional bf16 copy for next GEMM
// one row (1024 elems) per 256-thread block
// ---------------------------------------------------------------------------
__global__ __launch_bounds__(256) void add_ln_kernel(
    const float* __restrict__ X, const float* __restrict__ Y,
    const float* __restrict__ gamma, const float* __restrict__ beta,
    float* __restrict__ out, bf16* __restrict__ out_bf) {
  __shared__ float red[256];
  const int row = blockIdx.x;
  const float* x = X + (size_t)row * Dm;
  const float* y = Y + (size_t)row * Dm;

  float vals[4];
  float s = 0.f;
#pragma unroll
  for (int i = 0; i < 4; ++i) {
    int c = threadIdx.x + i * 256;
    vals[i] = x[c] + y[c];
    s += vals[i];
  }
  red[threadIdx.x] = s;
  __syncthreads();
  for (int d = 128; d > 0; d >>= 1) {
    if (threadIdx.x < d) red[threadIdx.x] += red[threadIdx.x + d];
    __syncthreads();
  }
  float mu = red[0] * (1.f / Dm);
  __syncthreads();

  float sq = 0.f;
#pragma unroll
  for (int i = 0; i < 4; ++i) { float t = vals[i] - mu; sq += t * t; }
  red[threadIdx.x] = sq;
  __syncthreads();
  for (int d = 128; d > 0; d >>= 1) {
    if (threadIdx.x < d) red[threadIdx.x] += red[threadIdx.x + d];
    __syncthreads();
  }
  float rstd = rsqrtf(red[0] * (1.f / Dm) + LN_EPSF);

#pragma unroll
  for (int i = 0; i < 4; ++i) {
    int c = threadIdx.x + i * 256;
    float hv = (vals[i] - mu) * rstd * gamma[c] + beta[c];
    out[(size_t)row * Dm + c] = hv;
    if (out_bf) out_bf[(size_t)row * Dm + c] = (bf16)hv;
  }
}

// ---------------------------------------------------------------------------
// launch
// ---------------------------------------------------------------------------
extern "C" void kernel_launch(void* const* d_in, const int* in_sizes, int n_in,
                              void* d_out, int out_size, void* d_ws,
                              size_t ws_size, hipStream_t stream) {
  (void)in_sizes; (void)n_in; (void)out_size; (void)ws_size;

  const float* x    = (const float*)d_in[0];
  const float* Wqkv = (const float*)d_in[1];
  const float* bqkv = (const float*)d_in[2];
  const float* Wout = (const float*)d_in[3];
  const float* bout = (const float*)d_in[4];
  const float* gma  = (const float*)d_in[5];
  const float* bta  = (const float*)d_in[6];
  const float* W1   = (const float*)d_in[7];
  const float* b1   = (const float*)d_in[8];
  const float* W2   = (const float*)d_in[9];
  const float* b2   = (const float*)d_in[10];

  const int M = Bn * Ln;  // 8192 token rows

  char* ws = (char*)d_ws;
  size_t off = 0;
  auto alloc = [&](size_t bytes) -> void* {
    void* p = ws + off;
    off = (off + bytes + 255) & ~(size_t)255;
    return p;
  };

  bf16*  xb    = (bf16*)alloc((size_t)M * Dm * 2);
  bf16*  wqkvb = (bf16*)alloc((size_t)Dm * 3 * Dm * 2);
  bf16*  woutb = (bf16*)alloc((size_t)Dm * Dm * 2);
  bf16*  w1b   = (bf16*)alloc((size_t)Dm * 4 * Dm * 2);
  bf16*  w2b   = (bf16*)alloc((size_t)4 * Dm * Dm * 2);
  bf16*  qkvb  = (bf16*)alloc((size_t)M * 3 * Dm * 2);      // 50.3 MB
  bf16*  attnb = (bf16*)alloc((size_t)M * Dm * 2);          // 16.8 MB
  float* y2    = (float*)alloc((size_t)M * Dm * 4);
  float* hbuf  = (float*)alloc((size_t)M * Dm * 4);
  bf16*  hb    = (bf16*)alloc((size_t)M * Dm * 2);
  // aliases: f1b reuses dead qkv+attn region (exactly 67.1MB); y4 reuses y2
  bf16*  f1b = qkvb;
  float* y4  = y2;

  auto cvt = [&](const float* in, bf16* o, size_t n) {
    int blocks = (int)((n + 256 * 8 - 1) / (256 * 8));
    f32_to_bf16_kernel<<<blocks, 256, 0, stream>>>(in, o, (int)n);
  };
  cvt(x,    xb,    (size_t)M * Dm);
  cvt(Wqkv, wqkvb, (size_t)Dm * 3 * Dm);
  cvt(Wout, woutb, (size_t)Dm * Dm);
  cvt(W1,   w1b,   (size_t)Dm * 4 * Dm);
  cvt(W2,   w2b,   (size_t)4 * Dm * Dm);

  dim3 blk(256);

  // 1) qkv = x @ Wqkv + bqkv  -> bf16
  gemm_bf16_kernel<true, false>
      <<<dim3(3 * Dm / 128, M / 128), blk, 0, stream>>>(
          xb, wqkvb, bqkv, qkvb, M, 3 * Dm, Dm);

  // 2) strided-head attention -> bf16 [M, Dm]
  hepos_attn_kernel<<<dim3(Ln / 128, NH, Bn), blk, 0, stream>>>(qkvb, attnb);

  // 3) y2 = attn @ Wout + bout -> f32
  gemm_bf16_kernel<false, false>
      <<<dim3(Dm / 128, M / 128), blk, 0, stream>>>(
          attnb, woutb, bout, y2, M, Dm, Dm);

  // 4) h = LN(x + y2) -> f32 + bf16
  add_ln_kernel<<<M, 256, 0, stream>>>(x, y2, gma, bta, hbuf, hb);

  // 5) f1 = relu(h @ W1 + b1) -> bf16
  gemm_bf16_kernel<true, true>
      <<<dim3(4 * Dm / 128, M / 128), blk, 0, stream>>>(
          hb, w1b, b1, f1b, M, 4 * Dm, Dm);

  // 6) y4 = f1 @ W2 + b2 -> f32
  gemm_bf16_kernel<false, false>
      <<<dim3(Dm / 128, M / 128), blk, 0, stream>>>(
          f1b, w2b, b2, y4, M, Dm, 4 * Dm);

  // 7) out = LN(h + y4) -> f32 d_out
  add_ln_kernel<<<M, 256, 0, stream>>>(hbuf, y4, gma, bta, (float*)d_out,
                                       nullptr);
}
